// TextEmbedding_11836929868626
// MI455X (gfx1250) — matrix operand order repaired and checked by reading, not compiled, not run on hardware
//
#include <hip/hip_runtime.h>
#include <hip/hip_bf16.h>
#include <math.h>

// Problem constants (match reference).
#define B_  16
#define L_  2048
#define D_  512
#define H_  1024
#define NL_ 4

typedef __attribute__((ext_vector_type(16))) _Float16 v16h;
typedef __attribute__((ext_vector_type(8)))  float    v8f;
typedef __attribute__((ext_vector_type(4)))  int      v4i_;

union FragU { unsigned int u[8]; v16h v; };
union Q4    { uint4 q; _Float16 h[8]; };

// ---------------------------------------------------------------------------
// CDNA5 async global->LDS staging (ASYNCcnt-tracked, VGPR-bypassing).
// Builtin signature (probe-verified): (v4i AS1*, v4i AS3*, Ii, Ii).
// ---------------------------------------------------------------------------
#if defined(__has_builtin)
#if __has_builtin(__builtin_amdgcn_global_load_async_to_lds_b128) && \
    __has_builtin(__builtin_amdgcn_s_wait_asynccnt)
#define ASYNC_LDS 1
#endif
#endif
#ifndef ASYNC_LDS
#define ASYNC_LDS 0
#endif

__device__ __forceinline__ void cp_b128(const _Float16* g, _Float16* l) {
#if ASYNC_LDS
  v4i_* gp = (v4i_*)(void*)g;   // strip const, generic typed pointer
  v4i_* lp = (v4i_*)l;
  __builtin_amdgcn_global_load_async_to_lds_b128(
      (__attribute__((address_space(1))) v4i_*)gp,
      (__attribute__((address_space(3))) v4i_*)lp, 0, 0);
#else
  *(uint4*)l = *(const uint4*)g;
#endif
}
__device__ __forceinline__ void wait_async_6() {   // two tiles may be in flight
#if ASYNC_LDS
  __builtin_amdgcn_s_wait_asynccnt(6);
#endif
}
__device__ __forceinline__ void wait_async_3() {   // one tile may be in flight
#if ASYNC_LDS
  __builtin_amdgcn_s_wait_asynccnt(3);
#endif
}
__device__ __forceinline__ void wait_async_0() {
#if ASYNC_LDS
  __builtin_amdgcn_s_wait_asynccnt(0);
#endif
}

// ---------------------------------------------------------------------------
// WMMA fragment gathers from LDS tiles stored K-contiguous per row.
// A (16x32 f16, ISA 7.12.2): lane holds M=lane&15; dword v holds pair
//   K0 = 2*(v&3) + (v>=4?16:0) + (lane>=16?8:0).
// B (32x16 f16): lane holds N=lane&15; dword v holds K0 = 2*(v&3) + (v>=4?16:0) + (lane>=16?8:0).
// Pairs are K-consecutive -> compiler coalesces into ds_load_b128 groups.
// ---------------------------------------------------------------------------
__device__ __forceinline__ v16h frag_A_row(const _Float16* row, int lane) {
  FragU f;
  const int hi = (lane >= 16) ? 8 : 0;
#pragma unroll
  for (int v = 0; v < 8; ++v) {
    const int k0 = 2 * (v & 3) + ((v >= 4) ? 16 : 0) + hi;
    f.u[v] = *(const unsigned int*)(row + k0);
  }
  return f.v;
}
__device__ __forceinline__ v16h frag_B_row(const _Float16* row, int lane) {
  FragU f;
  const int hi = (lane >= 16) ? 8 : 0;
#pragma unroll
  for (int v = 0; v < 8; ++v) {
    const int k0 = 2 * (v & 3) + ((v >= 4) ? 16 : 0) + hi;
    f.u[v] = *(const unsigned int*)(row + k0);
  }
  return f.v;
}

// ---------------------------------------------------------------------------
// Shared WMMA main loop: block tile 64(M) x 128(N), 8 waves (4x2), each wave
// 16x64 via 4 accumulators. 4-deep LDS ring, async staging with prefetch
// distance 2 (s_wait_asynccnt 6), a single barrier per K-step (the ring depth
// guarantees the overwritten buffer was last read two barriers ago).
// Fully unrolled (K is compile-time) so accumulators stay in place (SSA).
// ---------------------------------------------------------------------------
template <int K>
__device__ __forceinline__ void wmma_mainloop(const _Float16* __restrict__ Ag,
                                              const _Float16* __restrict__ Bg,
                                              _Float16* As, _Float16* Bs,
                                              int tid, int lane, int wm, int wn,
                                              v8f* acc) {
  constexpr int NT = K / 32;
  const int ar = tid >> 2, ac = (tid & 3) * 8;  // A: 64x32 tile, 1 x b128/thread
  const int bn = tid >> 1, bc = (tid & 1) * 16; // B: 128x32 tile, 2 x b128/thread
  const size_t aoff = (size_t)ar * K + ac;
  const size_t boff = (size_t)bn * K + bc;
#define STAGE_TILE(kt)                                                         \
  {                                                                            \
    const int buf_ = (kt) & 3;                                                 \
    cp_b128(Ag + aoff + (kt) * 32, As + buf_ * (64 * 32) + ar * 32 + ac);      \
    cp_b128(Bg + boff + (kt) * 32, Bs + buf_ * (128 * 32) + bn * 32 + bc);     \
    cp_b128(Bg + boff + (kt) * 32 + 8,                                         \
            Bs + buf_ * (128 * 32) + bn * 32 + bc + 8);                        \
  }
  STAGE_TILE(0);
  STAGE_TILE(1);
#pragma unroll
  for (int kt = 0; kt < NT; ++kt) {
    if (kt + 2 < NT)      { STAGE_TILE(kt + 2); wait_async_6(); }
    else if (kt + 1 < NT) { wait_async_3(); }
    else                  { wait_async_0(); }
    __syncthreads();
    const int buf = kt & 3;
    const v16h a =
        frag_A_row(As + buf * (64 * 32) + (wm * 16 + (lane & 15)) * 32, lane);
#pragma unroll
    for (int an = 0; an < 4; ++an) {
      const v16h b = frag_B_row(
          Bs + buf * (128 * 32) + (wn * 64 + an * 16 + (lane & 15)) * 32, lane);
      acc[an] = __builtin_amdgcn_wmma_f32_16x16x32_f16(false, a, false, b,
                                                       (short)0, acc[an], false,
                                                       false);
    }
  }
#undef STAGE_TILE
}

// ---------------------------------------------------------------------------
// Embedding + recomputed FREQS_CIS + padding mask -> f32 x[B,L,D]
// ---------------------------------------------------------------------------
__global__ __launch_bounds__(256)
void embed_kernel(const int* __restrict__ text, const float* __restrict__ table,
                  float* __restrict__ X) {
  const int bl  = blockIdx.x;
  const int l   = bl % L_;
  const int tok = text[bl] + 1;           // pad id -1 -> 0
#pragma unroll
  for (int e = 0; e < 2; ++e) {
    const int d = threadIdx.x + e * 256;
    float out = 0.f;
    if (tok != 0) {
      const int   i   = d & 255;
      const float f   = expf(-(float)i * (9.210340371976184f / 256.f)); // 10000^{-i/256}
      const float ang = (float)l * f;
      const float pe  = (d < 256) ? cosf(ang) : sinf(ang);
      out = table[(size_t)tok * D_ + d] + pe;
    }
    X[(size_t)bl * D_ + d] = out;
  }
}

// ---------------------------------------------------------------------------
// Weight prep: f32 [K][N] -> f16 transposed [N][K] via LDS-tiled transpose.
// ---------------------------------------------------------------------------
__global__ __launch_bounds__(256)
void transpose_f32f16(const float* __restrict__ W, _Float16* __restrict__ WT,
                      int K, int N) {
  __shared__ float tile[32][33];
  const int kB = blockIdx.y * 32, nB = blockIdx.x * 32;
  const int tx = threadIdx.x & 31, ty = threadIdx.x >> 5;   // 32x8
#pragma unroll
  for (int j = 0; j < 4; ++j)
    tile[ty + j * 8][tx] = W[(size_t)(kB + ty + j * 8) * N + nB + tx];
  __syncthreads();
#pragma unroll
  for (int j = 0; j < 4; ++j)
    WT[(size_t)(nB + ty + j * 8) * K + kB + tx] = (_Float16)tile[tx][ty + j * 8];
}

// ---------------------------------------------------------------------------
// Fused depthwise conv (k=7, pad 3) + LayerNorm over D -> f16 GEMM1 A-operand
// ---------------------------------------------------------------------------
__global__ __launch_bounds__(256)
void conv_ln_kernel(const float* __restrict__ X, const float* __restrict__ dw_w,
                    const float* __restrict__ dw_b, const float* __restrict__ ln_g,
                    const float* __restrict__ ln_b, _Float16* __restrict__ Yh,
                    int layer) {
  __shared__ float s_sum[256], s_sq[256];
  const int tid = threadIdx.x;
  const int bl  = blockIdx.x;
  const int b   = bl / L_, l = bl % L_;
  float v[2];
#pragma unroll
  for (int e = 0; e < 2; ++e) {
    const int d = tid + e * 256;
    float acc = dw_b[layer * D_ + d];
#pragma unroll
    for (int k = 0; k < 7; ++k) {
      const int ll = l + k - 3;
      if (ll >= 0 && ll < L_)
        acc += dw_w[(layer * D_ + d) * 7 + k] * X[((size_t)b * L_ + ll) * D_ + d];
    }
    v[e] = acc;
  }
  s_sum[tid] = v[0] + v[1];
  s_sq[tid]  = v[0] * v[0] + v[1] * v[1];
  __syncthreads();
  for (int off = 128; off > 0; off >>= 1) {
    if (tid < off) { s_sum[tid] += s_sum[tid + off]; s_sq[tid] += s_sq[tid + off]; }
    __syncthreads();
  }
  const float mean = s_sum[0] * (1.f / (float)D_);
  const float var  = s_sq[0] * (1.f / (float)D_) - mean * mean;
  const float rstd = rsqrtf(var + 1e-6f);
#pragma unroll
  for (int e = 0; e < 2; ++e) {
    const int d = tid + e * 256;
    const float y = (v[e] - mean) * rstd * ln_g[layer * D_ + d] + ln_b[layer * D_ + d];
    Yh[(size_t)bl * D_ + d] = (_Float16)y;
  }
}

// ---------------------------------------------------------------------------
// GEMM1: z = gelu(Y @ W1^T + b1); fold per-(b,h) sum(z^2) into Gsq for GRN.
// ---------------------------------------------------------------------------
__global__ __launch_bounds__(256)
void gemm1_gelu_grnsq(const _Float16* __restrict__ Yh, const _Float16* __restrict__ W1t,
                      const float* __restrict__ b1, _Float16* __restrict__ Zh,
                      float* __restrict__ Gsq, int layer) {
  __shared__ _Float16 As[4][64 * 32];
  __shared__ _Float16 Bs[4][128 * 32];
  __shared__ float colsq[128];
  const int tid = threadIdx.x, lane = tid & 31, wave = tid >> 5;
  const int wm = wave & 3, wn = wave >> 2;
  const int nBase = blockIdx.x * 128;
  const int rowBase = blockIdx.y * 64;          // 64 | L_ -> single batch b
  v8f zero8 = {0.f, 0.f, 0.f, 0.f, 0.f, 0.f, 0.f, 0.f};
  v8f acc[4]; acc[0] = zero8; acc[1] = zero8; acc[2] = zero8; acc[3] = zero8;

  wmma_mainloop<D_>(Yh + (size_t)rowBase * D_, W1t + (size_t)nBase * D_,
                    &As[0][0], &Bs[0][0], tid, lane, wm, wn, acc);

  // Epilogue: bias + exact GELU, store f16 z, fold column sum(z^2) into Gsq.
  if (tid < 128) colsq[tid] = 0.f;
  __syncthreads();
  const int bb = rowBase / L_;
#pragma unroll
  for (int an = 0; an < 4; ++an) {
    const int nl = wn * 64 + an * 16 + (lane & 15);
    const int n  = nBase + nl;
    const float bias = b1[layer * H_ + n];
    float sq = 0.f;
#pragma unroll
    for (int r = 0; r < 8; ++r) {
      const int m = rowBase + wm * 16 + ((lane < 16) ? r : r + 8);
      const float v = acc[an][r] + bias;
      const float g = 0.5f * v * (1.0f + erff(v * 0.70710678118f));
      Zh[(size_t)m * H_ + n] = (_Float16)g;
      sq += g * g;
    }
    atomicAdd(&colsq[nl], sq);
  }
  __syncthreads();
  if (tid < 128) atomicAdd(&Gsq[bb * H_ + nBase + tid], colsq[tid]);
}

// GRN normalization: Nx[b,h] = sqrt(Gsq)/(mean_h sqrt(Gsq) + 1e-6)
__global__ __launch_bounds__(256)
void grn_norm_kernel(const float* __restrict__ Gsq, float* __restrict__ Nx) {
  __shared__ float red[256];
  const int b = blockIdx.x, tid = threadIdx.x;
  float gx[4], s = 0.f;
#pragma unroll
  for (int j = 0; j < 4; ++j) { gx[j] = sqrtf(Gsq[b * H_ + tid + j * 256]); s += gx[j]; }
  red[tid] = s;
  __syncthreads();
  for (int off = 128; off > 0; off >>= 1) {
    if (tid < off) red[tid] += red[tid + off];
    __syncthreads();
  }
  const float denom = red[0] * (1.f / (float)H_) + 1e-6f;
#pragma unroll
  for (int j = 0; j < 4; ++j) Nx[b * H_ + tid + j * 256] = gx[j] / denom;
}

// GRN apply (in place): z <- z*(gg*Nx + 1) + gb. Keeps GEMM2 staging a pure copy.
__global__ __launch_bounds__(256)
void grn_apply_kernel(_Float16* __restrict__ Z, const float* __restrict__ gg,
                      const float* __restrict__ gb, const float* __restrict__ Nx,
                      int layer) {
  const size_t i8 = ((size_t)blockIdx.x * 256 + threadIdx.x) * 8;
  const int h = (int)(i8 % H_);
  const int b = (int)(i8 / ((size_t)L_ * H_));
  Q4 z; z.q = *(const uint4*)(Z + i8);
#pragma unroll
  for (int j = 0; j < 8; ++j) {
    const float s = gg[layer * H_ + h + j] * Nx[b * H_ + h + j] + 1.f;
    z.h[j] = (_Float16)((float)z.h[j] * s + gb[layer * H_ + h + j]);
  }
  *(uint4*)(Z + i8) = z.q;
}

// ---------------------------------------------------------------------------
// GEMM2: x = mask( z' @ W2^T + b2 + residual )
// ---------------------------------------------------------------------------
__global__ __launch_bounds__(256)
void gemm2_residual(const _Float16* __restrict__ Zh, const _Float16* __restrict__ W2t,
                    const float* __restrict__ b2, float* __restrict__ X,
                    const int* __restrict__ text, int layer) {
  __shared__ _Float16 As[4][64 * 32];
  __shared__ _Float16 Bs[4][128 * 32];
  const int tid = threadIdx.x, lane = tid & 31, wave = tid >> 5;
  const int wm = wave & 3, wn = wave >> 2;
  const int nBase = blockIdx.x * 128;
  const int rowBase = blockIdx.y * 64;
  v8f zero8 = {0.f, 0.f, 0.f, 0.f, 0.f, 0.f, 0.f, 0.f};
  v8f acc[4]; acc[0] = zero8; acc[1] = zero8; acc[2] = zero8; acc[3] = zero8;

  wmma_mainloop<H_>(Zh + (size_t)rowBase * H_, W2t + (size_t)nBase * H_,
                    &As[0][0], &Bs[0][0], tid, lane, wm, wn, acc);

  // Epilogue: +bias +residual, zero padded rows, write new residual stream.
#pragma unroll
  for (int an = 0; an < 4; ++an) {
    const int n = nBase + wn * 64 + an * 16 + (lane & 15);
    const float bias = b2[layer * D_ + n];
#pragma unroll
    for (int r = 0; r < 8; ++r) {
      const int m = rowBase + wm * 16 + ((lane < 16) ? r : r + 8);
      const bool masked = (text[m] == -1);       // text is flat [B*L]
      const float v = acc[an][r] + bias + X[(size_t)m * D_ + n];
      X[(size_t)m * D_ + n] = masked ? 0.f : v;
    }
  }
}

// ---------------------------------------------------------------------------
// Variable-length average upsample (repeat-interleave gather).
// ---------------------------------------------------------------------------
__global__ __launch_bounds__(256)
void upsample_kernel(const float* __restrict__ X, const int* __restrict__ text,
                     const int* __restrict__ audio, float* __restrict__ out) {
  __shared__ int pos_s[L_];
  __shared__ int src_s[L_];
  __shared__ int s_alen, s_vlen;
  const int b = blockIdx.y, slice = blockIdx.x, tid = threadIdx.x;
  for (int p = tid; p < L_; p += 256) pos_s[p] = L_;
  __syncthreads();
  if (tid == 0) {
    int cnt = 0, alen = 0;
    for (int l = 0; l < L_; ++l) {
      const int am = (audio[b * L_ + l] != 0);
      const int tv = (text[b * L_ + l] != -1);
      alen += am;
      if (am && tv) pos_s[cnt++] = l;
    }
    s_alen = alen; s_vlen = cnt;
  }
  __syncthreads();
  const int alen = s_alen, vlen = s_vlen;
  const int vl   = vlen > 1 ? vlen : 1;
  const int base = alen / vl, rem = alen % vl;
  const int cut  = (vl - rem) * base;
  const int bdiv = base > 1 ? base : 1;
  for (int p = tid; p < L_; p += 256) {
    int j = (p < cut) ? (p / bdiv) : ((vl - rem) + (p - cut) / (base + 1));
    if (j > L_ - 1) j = L_ - 1;
    int src = pos_s[j];
    if (src > L_ - 1) src = L_ - 1;
    const bool keep = (p < alen) && (vlen > 0);
    src_s[p] = keep ? src : -1;
  }
  __syncthreads();
  const int chunk = (L_ * D_) / 16;            // gridDim.x == 16 slices
  for (int idx = slice * chunk + tid; idx < (slice + 1) * chunk; idx += 256) {
    const int p = idx >> 9, d = idx & (D_ - 1);
    const int s = src_s[p];
    out[((size_t)b * L_ + p) * D_ + d] = (s < 0) ? 0.f
        : X[((size_t)b * L_ + s) * D_ + d];
  }
}

__global__ void zero_f32(float* p, int n) {
  const int i = blockIdx.x * 256 + threadIdx.x;
  if (i < n) p[i] = 0.f;
}

// ---------------------------------------------------------------------------
// Host-side orchestration (graph-capture safe: only kernel launches on stream)
// ---------------------------------------------------------------------------
extern "C" void kernel_launch(void* const* d_in, const int* in_sizes, int n_in,
                              void* d_out, int out_size, void* d_ws, size_t ws_size,
                              hipStream_t stream) {
  (void)in_sizes; (void)n_in; (void)out_size; (void)ws_size;
  const int*   text  = (const int*)d_in[0];
  const int*   audio = (const int*)d_in[1];
  // d_in[2] = seq_len scalar (== L_)
  const float* table = (const float*)d_in[3];
  const float* dw_w  = (const float*)d_in[4];
  const float* dw_b  = (const float*)d_in[5];
  const float* ln_g  = (const float*)d_in[6];
  const float* ln_b  = (const float*)d_in[7];
  const float* pw1_w = (const float*)d_in[8];
  const float* pw1_b = (const float*)d_in[9];
  const float* grn_g = (const float*)d_in[10];
  const float* grn_b = (const float*)d_in[11];
  const float* pw2_w = (const float*)d_in[12];
  const float* pw2_b = (const float*)d_in[13];

  char* ws = (char*)d_ws;
  size_t off = 0;
  float*    xbuf = (float*)(ws + off);    off += (size_t)B_ * L_ * D_ * 4;  // residual stream
  _Float16* yf16 = (_Float16*)(ws + off); off += (size_t)B_ * L_ * D_ * 2;  // LN output (GEMM1 A)
  _Float16* zf16 = (_Float16*)(ws + off); off += (size_t)B_ * L_ * H_ * 2;  // GELU/GRN (GEMM2 A)
  _Float16* w1t  = (_Float16*)(ws + off); off += (size_t)D_ * H_ * 2;       // W1^T [H][D] f16
  _Float16* w2t  = (_Float16*)(ws + off); off += (size_t)H_ * D_ * 2;       // W2^T [D][H] f16
  float*    Gsq  = (float*)(ws + off);    off += (size_t)B_ * H_ * 4;
  float*    Nx   = (float*)(ws + off);    off += (size_t)B_ * H_ * 4;

  embed_kernel<<<B_ * L_, 256, 0, stream>>>(text, table, xbuf);
  for (int layer = 0; layer < NL_; ++layer) {
    transpose_f32f16<<<dim3(H_ / 32, D_ / 32), 256, 0, stream>>>(
        pw1_w + (size_t)layer * D_ * H_, w1t, D_, H_);
    transpose_f32f16<<<dim3(D_ / 32, H_ / 32), 256, 0, stream>>>(
        pw2_w + (size_t)layer * H_ * D_, w2t, H_, D_);
    zero_f32<<<(B_ * H_) / 256, 256, 0, stream>>>(Gsq, B_ * H_);
    conv_ln_kernel<<<B_ * L_, 256, 0, stream>>>(xbuf, dw_w, dw_b, ln_g, ln_b, yf16, layer);
    gemm1_gelu_grnsq<<<dim3(H_ / 128, (B_ * L_) / 64), 256, 0, stream>>>(
        yf16, w1t, pw1_b, zf16, Gsq, layer);
    grn_norm_kernel<<<B_, 256, 0, stream>>>(Gsq, Nx);
    grn_apply_kernel<<<(B_ * L_ * H_) / (256 * 8), 256, 0, stream>>>(
        zf16, grn_g, grn_b, Nx, layer);
    gemm2_residual<<<dim3(D_ / 128, (B_ * L_) / 64), 256, 0, stream>>>(
        zf16, w2t, pw2_b, xbuf, text, layer);
  }
  upsample_kernel<<<dim3(16, B_), 256, 0, stream>>>(xbuf, text, audio, (float*)d_out);
}
